// Memory_2654289789385
// MI455X (gfx1250) — matrix operand order repaired and checked by reading, not compiled, not run on hardware
//
#include <hip/hip_runtime.h>
#include <hip/hip_bf16.h>
#include <stdint.h>

// Problem dimensions (from reference): query [64,256,512], keys [2048,512]
#define NQ      16384   // 64*256 query rows
#define DDIM    512
#define MSLOTS  2048
#define RT      128     // rows per workgroup in score kernel
#define CT      128     // cols per tile
#define KT      32      // K per WMMA step (bf16 16x16x32)
#define TEMP_UPDATE 1e-5f
#define NORM_EPS    1e-12f

typedef __attribute__((ext_vector_type(16))) __bf16 v16bf;
typedef __attribute__((ext_vector_type(8)))  float  v8f;

union AFrag { v16bf v; uint4 u[2]; };

// ---- float ordering helpers (monotone f32 <-> u32 map for atomicMax) ----
__device__ __forceinline__ unsigned ford(float f) {
    unsigned b = __float_as_uint(f);
    return (b & 0x80000000u) ? ~b : (b | 0x80000000u);
}
__device__ __forceinline__ float fdec(unsigned o) {
    unsigned b = (o & 0x80000000u) ? (o ^ 0x80000000u) : ~o;
    return __uint_as_float(b);
}

// ---- CDNA5 data movers -------------------------------------------------
#if defined(__AMDGCN__)
typedef int          v4i_vs __attribute__((vector_size(16)));
typedef unsigned int v4u_vs __attribute__((vector_size(16)));
typedef int          v8i_vs __attribute__((vector_size(32)));
typedef __attribute__((address_space(1))) v4i_vs* as1_v4i_ptr;  // global int4*
typedef __attribute__((address_space(3))) v4i_vs* as3_v4i_ptr;  // LDS int4*

#if __has_builtin(__builtin_amdgcn_tensor_load_to_lds) && \
    __has_builtin(__builtin_amdgcn_s_wait_tensorcnt)
#define USE_TDM 1
#endif
#if __has_include(<hip/amd_detail/amd_gfx1250_TDM.h>)
#define TDM_SIX_ARGS 1   // therock/clang-23 header toolchain uses 6-arg builtin
#endif
#endif

#if defined(USE_TDM)
// Issue one TDM DMA of a [CT rows x KT cols] bf16 tile (row stride DDIM
// elements) from global memory into contiguous LDS. Wave-uniform args.
__device__ __forceinline__ void tdm_load_tile(const __bf16* gsrc, unsigned lds_off) {
    unsigned long long ga = (unsigned long long)(uintptr_t)gsrc;
    // D# group 0: count=1 | lds_addr | global_addr[56:0] | type=2
    v4u_vs g0 = { 1u,
                  lds_off,
                  (unsigned)(ga & 0xFFFFFFFFu),
                  (unsigned)((ga >> 32) & 0x01FFFFFFu) | (2u << 30) };
    // D# group 1: data_size=2B; tensor_dim0=DDIM; tensor_dim1=MSLOTS;
    // tile_dim0=KT (contiguous K), tile_dim1=CT (columns); dim0 stride=DDIM
    v8i_vs g1 = { (int)(1u << 16),            // [17:16] data_size = 1 (2 bytes)
                  (int)((unsigned)DDIM << 16),   // [63:48]  tensor_dim0 lo16
                  (int)((unsigned)MSLOTS << 16), // [95:80]  tensor_dim1 lo16
                  (int)((unsigned)KT << 16),     // [127:112] tile_dim0
                  (int)CT,                        // [143:128] tile_dim1
                  (int)DDIM,                      // [191:160] tensor_dim0_stride
                  0, 0 };
    v4i_vs gz = { 0, 0, 0, 0 };                  // groups 2/3 unused (2-D tile)
#if defined(TDM_SIX_ARGS)
    v8i_vs gz8 = { 0, 0, 0, 0, 0, 0, 0, 0 };
    __builtin_amdgcn_tensor_load_to_lds(g0, g1, gz, gz, gz8, 0);
#else
    __builtin_amdgcn_tensor_load_to_lds(g0, g1, gz, gz, 0);
#endif
}
#endif

__device__ __forceinline__ void async_copy_b128(const __bf16* gsrc, __bf16* ldst) {
#if defined(__AMDGCN__)
#if __has_builtin(__builtin_amdgcn_global_load_async_to_lds_b128)
    __builtin_amdgcn_global_load_async_to_lds_b128(
        (as1_v4i_ptr)(uintptr_t)gsrc,
        (as3_v4i_ptr)(uintptr_t)ldst,   // low 32 bits of flat = LDS offset
        0, 0);
#elif defined(__gfx1250__)
    unsigned loff = (unsigned)(uintptr_t)ldst;
    asm volatile("global_load_async_to_lds_b128 %0, %1, off"
                 :: "v"(loff), "v"(gsrc) : "memory");
#endif
#else
    (void)gsrc; (void)ldst;
#endif
}

__device__ __forceinline__ void wait_async0() {
#if defined(__AMDGCN__)
#if __has_builtin(__builtin_amdgcn_s_wait_asynccnt)
    __builtin_amdgcn_s_wait_asynccnt(0);
#elif defined(__gfx1250__)
    asm volatile("s_wait_asynccnt 0x0" ::: "memory");
#endif
#endif
}

// ---- K0: keys -> bf16, zero update accumulator, init column-max ----
__global__ __launch_bounds__(256) void prep_kernel(
    const float* __restrict__ keys, __bf16* __restrict__ kb,
    float* __restrict__ upd, unsigned* __restrict__ colmax_ord)
{
    int i = blockIdx.x * 256 + threadIdx.x;   // grid covers MSLOTS*DDIM
    kb[i]  = (__bf16)keys[i];
    upd[i] = 0.0f;
    if (i < MSLOTS) colmax_ord[i] = 0u;       // ord(x) > 0 for all finite x
}

// ---- K1: L2-normalize each query row, emit bf16 copy for WMMA ----
__global__ __launch_bounds__(256) void qnorm_kernel(
    const float* __restrict__ query, __bf16* __restrict__ qb)
{
    __shared__ float sred[8];
    const int n   = blockIdx.x;
    const int tid = threadIdx.x;
    const float* qrow = query + (size_t)n * DDIM;
    float a0 = qrow[tid], a1 = qrow[tid + 256];
    float ss = a0 * a0 + a1 * a1;
    #pragma unroll
    for (int d = 16; d >= 1; d >>= 1) ss += __shfl_xor(ss, d);
    if ((tid & 31) == 0) sred[tid >> 5] = ss;
    __syncthreads();
    float tot = sred[0];
    #pragma unroll
    for (int w = 1; w < 8; ++w) tot += sred[w];
    float inv = 1.0f / fmaxf(sqrtf(tot), NORM_EPS);
    __bf16* dst = qb + (size_t)n * DDIM;
    dst[tid]       = (__bf16)(a0 * inv);
    dst[tid + 256] = (__bf16)(a1 * inv);
}

// ---- K2: fused score GEMM (bf16 WMMA) + row max/argmax + column max ----
// Grid: NQ/RT = 128 workgroups of 256 threads (8 waves). Each WG owns 128
// rows and sweeps all 2048 columns; keys tiles are double-buffered in LDS
// and staged by the Tensor Data Mover (one DMA per tile, TENSORcnt) or, if
// the TDM builtin is unavailable, by async global->LDS copies (ASYNCcnt).
__global__ __launch_bounds__(256) void score_kernel(
    const __bf16* __restrict__ qb, const __bf16* __restrict__ kb,
    float* __restrict__ row_max, unsigned* __restrict__ row_arg,
    unsigned* __restrict__ colmax_ord)
{
    __shared__ alignas(32) __bf16 sK[2][CT * KT];   // 2 x 8 KB keys tiles
    __shared__ float sCol[8][CT];                   // 4 KB column partials

    const int tid  = threadIdx.x;
    const int wv   = tid >> 5;       // wave 0..7 -> 16-row strip
    const int lane = tid & 31;
    const int h    = lane >> 4;      // lane half (selects K chunk / M half)
    const int ln   = lane & 15;
    const int r0   = blockIdx.x * RT;
    const int rowbase = r0 + wv * 16 + 8 * h;       // C rows: rowbase + r

    // staging assignment (async fallback): thread -> (column, chunk)
    const int scol = tid >> 1;
    const int sch  = (tid & 1) * 16;

    float bestv[8];
    int   bestc[8];
    #pragma unroll
    for (int r = 0; r < 8; ++r) { bestv[r] = -3.4e38f; bestc[r] = 0; }

    for (int ct = 0; ct < MSLOTS / CT; ++ct) {
        const int c0 = ct * CT;
        v8f acc[8];
        #pragma unroll
        for (int f = 0; f < 8; ++f) {
            v8f z = {0.f, 0.f, 0.f, 0.f, 0.f, 0.f, 0.f, 0.f};
            acc[f] = z;
        }

        // stage first K tile into buffer 0
#if defined(USE_TDM)
        if (wv == 0)
            tdm_load_tile(kb + (size_t)c0 * DDIM, (unsigned)(uintptr_t)&sK[0][0]);
#else
        {
            const __bf16* src = kb + (size_t)(c0 + scol) * DDIM + sch;
            async_copy_b128(src,     &sK[0][scol * KT + sch]);
            async_copy_b128(src + 8, &sK[0][scol * KT + sch + 8]);
        }
#endif

        #pragma unroll
        for (int ks = 0; ks < DDIM / KT; ++ks) {
            const int k0  = ks * KT;
            const int buf = ks & 1;
#if defined(USE_TDM)
            if (wv == 0) __builtin_amdgcn_s_wait_tensorcnt(0);  // DMA done
#else
            wait_async0();       // my tile-k loads done
#endif
            __syncthreads();     // tile k visible to all waves

            if (k0 + KT < DDIM) {   // stage tile k+1 into the other buffer
#if defined(USE_TDM)
                if (wv == 0)
                    tdm_load_tile(kb + (size_t)c0 * DDIM + (k0 + KT),
                                  (unsigned)(uintptr_t)&sK[buf ^ 1][0]);
#else
                const __bf16* src = kb + (size_t)(c0 + scol) * DDIM + (k0 + KT) + sch;
                async_copy_b128(src,     &sK[buf ^ 1][scol * KT + sch]);
                async_copy_b128(src + 8, &sK[buf ^ 1][scol * KT + sch + 8]);
                if (k0 + 2 * KT < DDIM)  // hint tile k+2 -> global_prefetch_b8
                    __builtin_prefetch(src + KT, 0, 0);
#endif
            }

            // A fragment: 16x32 bf16, row = strip+ln; lanes 0-15 hold K[0..16),
            // lanes 16-31 hold K[16..32)
            AFrag af;
            const __bf16* qrow = qb + (size_t)(r0 + wv * 16 + ln) * DDIM + k0;
            af.u[0] = *reinterpret_cast<const uint4*>(qrow + 16 * h);
            af.u[1] = *reinterpret_cast<const uint4*>(qrow + 16 * h + 8);

            #pragma unroll
            for (int f = 0; f < 8; ++f) {
                // B fragment: 32x16 bf16, lane holds col (16f+ln), K chunk 16h
                v16bf bfrag = *reinterpret_cast<const v16bf*>(
                    &sK[buf][(16 * f + ln) * KT + 16 * h]);
                acc[f] = __builtin_amdgcn_wmma_f32_16x16x32_bf16(
                    false, af.v, false, bfrag, (short)0, acc[f], false, false);
            }
        }

        // running per-row max/argmax (row = rowbase + r, col = c0+16f+ln)
        #pragma unroll
        for (int f = 0; f < 8; ++f) {
            #pragma unroll
            for (int r = 0; r < 8; ++r) {
                float v = acc[f][r];
                int   c = c0 + 16 * f + ln;
                if (v > bestv[r]) { bestv[r] = v; bestc[r] = c; }
            }
        }

        // per-column partial max for this tile
        #pragma unroll
        for (int f = 0; f < 8; ++f) {
            float cm = acc[f][0];
            #pragma unroll
            for (int r = 1; r < 8; ++r) cm = fmaxf(cm, acc[f][r]);
            cm = fmaxf(cm, __shfl_xor(cm, 16));   // combine both 8-row halves
            if (h == 0) sCol[wv][16 * f + ln] = cm;
        }
        __syncthreads();
        if (tid < CT) {
            float m = sCol[0][tid];
            #pragma unroll
            for (int w = 1; w < 8; ++w) m = fmaxf(m, sCol[w][tid]);
            atomicMax(&colmax_ord[c0 + tid], ford(m));
        }
        // the wait+barrier at ks=0 of the next tile protects sCol reuse;
        // the next stage targets buffer 0 while the last reads hit buffer 1
    }

    // reduce row best across the 16 lanes that share each row
    #pragma unroll
    for (int r = 0; r < 8; ++r) {
        float v = bestv[r];
        int   c = bestc[r];
        #pragma unroll
        for (int d = 1; d < 16; d <<= 1) {
            float ov = __shfl_xor(v, d);
            int   oc = __shfl_xor(c, d);
            if (ov > v || (ov == v && oc < c)) { v = ov; c = oc; }
        }
        if (ln == 0) {
            row_max[rowbase + r] = v;
            row_arg[rowbase + r] = (unsigned)c;
        }
    }
}

// ---- K3: per-row weight w = exp(rowmax - colmax[g]); scatter w*qn into upd ----
__global__ __launch_bounds__(256) void scatter_kernel(
    const float* __restrict__ query, const float* __restrict__ row_max,
    const unsigned* __restrict__ row_arg, const unsigned* __restrict__ colmax_ord,
    float* __restrict__ upd)
{
    __shared__ float sred[8];
    const int n   = blockIdx.x;
    const int tid = threadIdx.x;
    const float* qrow = query + (size_t)n * DDIM;
    float a0 = qrow[tid], a1 = qrow[tid + 256];
    float ss = a0 * a0 + a1 * a1;
    #pragma unroll
    for (int d = 16; d >= 1; d >>= 1) ss += __shfl_xor(ss, d);
    if ((tid & 31) == 0) sred[tid >> 5] = ss;
    __syncthreads();
    float tot = sred[0];
    #pragma unroll
    for (int w = 1; w < 8; ++w) tot += sred[w];
    float inv = 1.0f / fmaxf(sqrtf(tot), NORM_EPS);

    unsigned g = row_arg[n];
    float w = __expf(row_max[n] - fdec(colmax_ord[g])) * inv;
    float* dst = upd + (size_t)g * DDIM;
    atomicAdd(&dst[tid],       w * a0);
    atomicAdd(&dst[tid + 256], w * a1);
}

// ---- K4: out = l2norm(keys + 1e-5 * upd) per slot ----
__global__ __launch_bounds__(256) void finalize_kernel(
    const float* __restrict__ keys, const float* __restrict__ upd,
    float* __restrict__ out)
{
    __shared__ float sred[8];
    const int j   = blockIdx.x;
    const int tid = threadIdx.x;
    const size_t base = (size_t)j * DDIM;
    float v0 = fmaf(TEMP_UPDATE, upd[base + tid],       keys[base + tid]);
    float v1 = fmaf(TEMP_UPDATE, upd[base + tid + 256], keys[base + tid + 256]);
    float ss = v0 * v0 + v1 * v1;
    #pragma unroll
    for (int d = 16; d >= 1; d >>= 1) ss += __shfl_xor(ss, d);
    if ((tid & 31) == 0) sred[tid >> 5] = ss;
    __syncthreads();
    float tot = sred[0];
    #pragma unroll
    for (int w = 1; w < 8; ++w) tot += sred[w];
    float inv = 1.0f / fmaxf(sqrtf(tot), NORM_EPS);
    out[base + tid]       = v0 * inv;
    out[base + tid + 256] = v1 * inv;
}

extern "C" void kernel_launch(void* const* d_in, const int* in_sizes, int n_in,
                              void* d_out, int out_size, void* d_ws, size_t ws_size,
                              hipStream_t stream) {
    (void)in_sizes; (void)n_in; (void)out_size; (void)ws_size;
    const float* query = (const float*)d_in[0];   // [64,256,512]
    const float* keys  = (const float*)d_in[1];   // [2048,512]
    float* out = (float*)d_out;                   // [2048,512]

    // workspace partition (~22.2 MiB total, all chunks 256B aligned)
    char* ws = (char*)d_ws;
    __bf16*   qb         = (__bf16*)ws;             ws += (size_t)NQ * DDIM * 2;
    __bf16*   kb         = (__bf16*)ws;             ws += (size_t)MSLOTS * DDIM * 2;
    float*    upd        = (float*)ws;              ws += (size_t)MSLOTS * DDIM * 4;
    float*    row_max    = (float*)ws;              ws += (size_t)NQ * 4;
    unsigned* row_arg    = (unsigned*)ws;           ws += (size_t)NQ * 4;
    unsigned* colmax_ord = (unsigned*)ws;           ws += (size_t)MSLOTS * 4;

    prep_kernel<<<(MSLOTS * DDIM) / 256, 256, 0, stream>>>(keys, kb, upd, colmax_ord);
    qnorm_kernel<<<NQ, 256, 0, stream>>>(query, qb);
    score_kernel<<<NQ / RT, 256, 0, stream>>>(qb, kb, row_max, row_arg, colmax_ord);
    scatter_kernel<<<NQ, 256, 0, stream>>>(query, row_max, row_arg, colmax_ord, upd);
    finalize_kernel<<<MSLOTS, 256, 0, stream>>>(keys, upd, out);
}
